// HeatMIM_49976239456923
// MI455X (gfx1250) — compile-verified, matrix operand order checked by reading, not run on hardware
//
#include <hip/hip_runtime.h>
#include <hip/hip_bf16.h>

// ---------------------------------------------------------------------------
// DGM forward (HeatMIM) on MI455X / gfx1250.
// Compute-bound (~327 GFLOP over ~25MB): v_wmma_f32_16x16x32_bf16 path.
// Round-3 changes vs round-2:
//   * weight staging via GLOBAL_LOAD_ASYNC_TO_LDS_B128 (ASYNCcnt-tracked DMA)
//     into a double-buffered LDS weight tile: no VGPR prefetch file, no
//     commit stores, one barrier per GEMM instead of two.
//   * input tile copy also uses the async global->LDS path.
// ---------------------------------------------------------------------------

#define HID   128
#define DIN   17
#define KIN   32      // padded K for the input layer
#define NDGM  3
#define MT    128     // rows per workgroup
#define SP    136     // padded LDS row stride (bf16 elems) -> bank-conflict free

#define TILE_B   (MT * SP * 2)        // bytes per LDS tile (34,816)
#define OFF_X    0
#define OFF_S    (1 * TILE_B)
#define OFF_T    (2 * TILE_B)
#define OFF_W0   (3 * TILE_B)
#define OFF_W1   (4 * TILE_B)

typedef __attribute__((ext_vector_type(16))) __bf16 v16bf_t;
typedef __attribute__((ext_vector_type(8)))  float  v8f_t;

union FragU {
    v16bf_t v;
    uint4   u[2];
};

static __device__ __forceinline__ unsigned short f32_to_bf16(float f) {
    union { float f; unsigned u; } v; v.f = f;
    unsigned r = v.u + 0x7FFFu + ((v.u >> 16) & 1u);   // round-to-nearest-even
    return (unsigned short)(r >> 16);
}
static __device__ __forceinline__ float bf16_to_f32(unsigned short h) {
    union { unsigned u; float f; } v; v.u = ((unsigned)h) << 16;
    return v.f;
}
// branchless: sigmoid(x) = rcp(1 + e^-x); correct limits via inf/0 semantics
static __device__ __forceinline__ float fast_sigmoid(float x) {
    return __builtin_amdgcn_rcpf(1.0f + __expf(-x));
}
// branchless: tanh(x) = 1 - 2*rcp(e^{2x} + 1);  x->+inf: 1, x->-inf: -1
static __device__ __forceinline__ float fast_tanh(float x) {
    return 1.0f - 2.0f * __builtin_amdgcn_rcpf(__expf(2.0f * x) + 1.0f);
}

// ---- CDNA5 async global->LDS DMA (tracked by ASYNCcnt) --------------------
static __device__ __forceinline__ void async_b128(unsigned lds_byte_off,
                                                  unsigned glb_byte_off,
                                                  const void* base) {
    asm volatile("global_load_async_to_lds_b128 %0, %1, %2"
                 :: "v"(lds_byte_off), "v"(glb_byte_off),
                    "s"((unsigned long long)base)
                 : "memory");
}
static __device__ __forceinline__ void async_wait_all() {
    asm volatile("s_wait_asynccnt 0x0" ::: "memory");
}

struct BranchW {
    const unsigned short* in_wT;  // bf16 [128][32]   (out-major, padded K)
    const unsigned short* h1T;    // bf16 [128][128]
    const unsigned short* h2T;    // bf16 [128][128]
    const unsigned short* dgmT;   // bf16 [3][12][128][128] (each out-major)
    const float* in_b;            // [128]
    const float* h1_b;            // [128]
    const float* h2_b;            // [128]
    const float* dgm_b;           // [3][12][128]
    const float* out_w;           // f32 (128, odim)
    const float* out_b;           // f32 (odim,)
    int odim;
};

// ------------------------- prep kernels ------------------------------------

__global__ void k_pack_input(const float* __restrict__ X, const float* __restrict__ T,
                             unsigned short* __restrict__ dst, int n) {
    int row = blockIdx.x * blockDim.x + threadIdx.x;
    if (row >= n) return;
    unsigned short* o = dst + (size_t)row * KIN;
    const float* xr = X + (size_t)row * 16;
#pragma unroll
    for (int d = 0; d < 16; ++d) o[d] = f32_to_bf16(xr[d]);
    o[16] = f32_to_bf16(T[row]);
#pragma unroll
    for (int d = 17; d < KIN; ++d) o[d] = 0;
}

// src (in, out) fp32 -> dst [out][K] bf16, zero-padded K (for the 17->128 layer)
__global__ void k_conv_pad(const float* __restrict__ src, unsigned short* __restrict__ dst,
                           int in, int out, int K) {
    int idx = blockIdx.x * blockDim.x + threadIdx.x;
    if (idx >= out * K) return;
    int o = idx / K, i = idx - o * K;
    dst[idx] = (i < in) ? f32_to_bf16(src[(size_t)i * out + o]) : (unsigned short)0;
}

// batched 128x128 transpose-convert: src[b](in,out) fp32 -> dst[b][out][in] bf16
__global__ void k_conv_sq(const float* __restrict__ src, unsigned short* __restrict__ dst,
                          int batch) {
    int idx = blockIdx.x * blockDim.x + threadIdx.x;
    if (idx >= batch * HID * HID) return;
    int b = idx >> 14;          // /16384
    int r = idx & 16383;
    int o = r >> 7, i = r & 127;
    dst[idx] = f32_to_bf16(src[(size_t)b * 16384 + (size_t)i * HID + o]);
}

// ------------------------- wave-level GEMM helpers -------------------------

// acc[nt] += A(16x32) x W(32x16) over KS K-steps. A, W: LDS bf16 tiles, stride SP.
// B-fragments batched 4-at-a-time so the ds_load waits overlap wmma issue.
template <int KS>
static __device__ __forceinline__ void wave_gemm_acc(const unsigned short* A,
                                                     const unsigned short* W,
                                                     v8f_t acc[8], int mr, int lane) {
    const int half = (lane >> 4) & 1;
    const int ml   = lane & 15;
#pragma unroll
    for (int ks = 0; ks < KS; ++ks) {
        const int k0 = ks * 32;
        FragU fa;
        const unsigned short* pa = A + (mr + ml) * SP + k0 + half * 8;
        fa.u[0] = *(const uint4*)(pa);
        fa.u[1] = *(const uint4*)(pa + 16);
#pragma unroll
        for (int h = 0; h < 2; ++h) {
            FragU fb[4];
#pragma unroll
            for (int q = 0; q < 4; ++q) {
                const int nt = h * 4 + q;
                const unsigned short* pb = W + (nt * 16 + ml) * SP + k0 + half * 8;
                fb[q].u[0] = *(const uint4*)(pb);
                fb[q].u[1] = *(const uint4*)(pb + 16);
            }
#pragma unroll
            for (int q = 0; q < 4; ++q) {
                const int nt = h * 4 + q;
                acc[nt] = __builtin_amdgcn_wmma_f32_16x16x32_bf16(
                    false, fa.v, false, fb[q].v, (short)0, acc[nt], false, false);
            }
        }
    }
}

// issue async DMA of a [128][K] bf16 weight matrix into LDS tile at wt_off
template <int K>
static __device__ __forceinline__ void async_stage_w(const unsigned short* __restrict__ g,
                                                     unsigned wt_off) {
    constexpr int CPR = K / 8;              // 16B chunks per row
    constexpr int NCH = HID * CPR / 256;    // chunks per thread (8 or 2)
#pragma unroll
    for (int j = 0; j < NCH; ++j) {
        const int c   = threadIdx.x + 256 * j;
        const int row = c / CPR, cc = c % CPR;
        async_b128(wt_off + (unsigned)((row * SP + cc * 8) * 2),
                   (unsigned)((row * K + cc * 8) * 2), g);
    }
}

static __device__ __forceinline__ void zero_acc(v8f_t acc[8]) {
#pragma unroll
    for (int nt = 0; nt < 8; ++nt) acc[nt] = (v8f_t){0,0,0,0,0,0,0,0};
}

// ACT: 0=none 1=tanh  -- store C-layout accumulators (+bias) to LDS bf16 tile
template <int ACT>
static __device__ __forceinline__ void store_act(unsigned short* dst, const v8f_t acc[8],
                                                 const float* __restrict__ b,
                                                 int mr, int lane) {
    const int half = (lane >> 4) & 1;
    const int ml   = lane & 15;
#pragma unroll
    for (int nt = 0; nt < 8; ++nt) {
        const int col  = nt * 16 + ml;
        const float bv = b[col];
#pragma unroll
        for (int r = 0; r < 8; ++r) {
            float v = acc[nt][r] + bv;
            if (ACT == 1) v = fast_tanh(v);
            dst[(mr + r + 8 * half) * SP + col] = f32_to_bf16(v);
        }
    }
}

// ------------------------- main fused branch kernel ------------------------

__global__ __launch_bounds__(256)
void dgm_branch_kernel(const float* __restrict__ X, const float* __restrict__ T,
                       const unsigned short* __restrict__ inp,
                       BranchW BU, BranchW BP, float* __restrict__ out, int n) {
    extern __shared__ char smem_raw[];
    unsigned short* bufX = (unsigned short*)(smem_raw + OFF_X);   // x (== S1)
    unsigned short* bufS = (unsigned short*)(smem_raw + OFF_S);   // S
    unsigned short* bufT = (unsigned short*)(smem_raw + OFF_T);   // input / S*R
    unsigned short* wts[2] = { (unsigned short*)(smem_raw + OFF_W0),
                               (unsigned short*)(smem_raw + OFF_W1) };
    const unsigned wto[2] = { OFF_W0, OFF_W1 };

    const int tid  = threadIdx.x;
    const int lane = tid & 31;
    const int wv   = tid >> 5;
    const int mr   = wv * 16;
    const int half = (lane >> 4) & 1;
    const int ml   = lane & 15;
    const int R0   = blockIdx.x * MT;

    const BranchW bw = (blockIdx.y == 0) ? BU : BP;

    v8f_t acc[8];
    int p = 0;   // wts[p] holds the weight for the *current* GEMM

    // --- async DMA: input rows -> bufT, in_w -> wt0 ---
#pragma unroll
    for (int j = 0; j < 2; ++j) {                 // 512 chunks of 16B
        const int c = tid + 256 * j;
        const int row = c >> 2, cc = c & 3;
        async_b128(OFF_T + (unsigned)((row * SP + cc * 8) * 2),
                   (unsigned)(((size_t)(R0 + row) * KIN + cc * 8) * 2), inp);
    }
    async_stage_w<KIN>(bw.in_wT, wto[0]);
    async_wait_all();
    __syncthreads();

    // --- feed-in pipeline: stage next weight via async DMA during each GEMM ---
    async_stage_w<HID>(bw.h1T, wto[1]);
    zero_acc(acc);
    wave_gemm_acc<KIN / 32>(bufT, wts[0], acc, mr, lane);
    store_act<1>(bufX, acc, bw.in_b, mr, lane);    // h = tanh(inp@in_w + b)
    async_wait_all(); __syncthreads(); p = 1;

    async_stage_w<HID>(bw.h2T, wto[0]);
    zero_acc(acc);
    wave_gemm_acc<4>(bufX, wts[1], acc, mr, lane);
    store_act<1>(bufT, acc, bw.h1_b, mr, lane);    // h = tanh(h@h1_w + b)
    async_wait_all(); __syncthreads(); p = 0;

    async_stage_w<HID>(bw.dgmT + (size_t)6 * HID * HID, wto[1]);   // W(l=0, 6)
    zero_acc(acc);
    wave_gemm_acc<4>(bufT, wts[0], acc, mr, lane);
    store_act<0>(bufX, acc, bw.h2_b, mr, lane);    // x = h@h2_w + b  (no act)
    async_wait_all(); __syncthreads(); p = 1;

    // --- S = x (cooperative LDS copy of the 128-col region) ---
    for (int c = tid; c < MT * (HID / 8); c += 256) {
        int row = c >> 4, cc = c & 15;
        *(uint4*)(bufS + row * SP + cc * 8) = *(const uint4*)(bufX + row * SP + cc * 8);
    }
    __syncthreads();

    // --- DGM stack (S1 == x == bufX for all layers) ---
#pragma unroll
    for (int l = 0; l < NDGM; ++l) {
        const unsigned short* Wd = bw.dgmT + (size_t)l * 12 * HID * HID;
        const float*          Bd = bw.dgm_b + (size_t)l * 12 * HID;
#define WW(j) (Wd + (size_t)(j) * HID * HID)
#define BB(j) (Bd + (size_t)(j) * HID)
        // stage NEXT weight into the other tile, GEMM on the current tile
#define GSTEP(A_, NEXTW_)                                     \
        do { async_stage_w<HID>((NEXTW_), wto[p ^ 1]);        \
             wave_gemm_acc<4>((A_), wts[p], acc, mr, lane); } while (0)
#define COMMIT() do { async_wait_all(); __syncthreads(); p ^= 1; } while (0)

        // ---- R = sigmoid(x@w6 + S@w7 + S1@w8 + b);  bufT = S * R ----
        zero_acc(acc);
        GSTEP(bufX, WW(7)); COMMIT();
        GSTEP(bufS, WW(8)); COMMIT();
        GSTEP(bufX, WW(0));
#pragma unroll
        for (int nt = 0; nt < 8; ++nt) {
            const int col = nt * 16 + ml;
            const float bv = BB(6)[col] + BB(7)[col] + BB(8)[col];
#pragma unroll
            for (int r = 0; r < 8; ++r) {
                const int row = mr + r + 8 * half;
                float rv = fast_sigmoid(acc[nt][r] + bv);
                float s  = bf16_to_f32(bufS[row * SP + col]);
                bufT[row * SP + col] = f32_to_bf16(rv * s);
            }
        }
        COMMIT();                                          // wts[p] = W(0)

        // ---- Z;  zs = Z * S (registers) ----
        v8f_t zs[8];
        zero_acc(acc);
        GSTEP(bufX, WW(1)); COMMIT();
        GSTEP(bufS, WW(2)); COMMIT();
        GSTEP(bufX, WW(3));
#pragma unroll
        for (int nt = 0; nt < 8; ++nt) {
            const int col = nt * 16 + ml;
            const float bv = BB(0)[col] + BB(1)[col] + BB(2)[col];
#pragma unroll
            for (int r = 0; r < 8; ++r) {
                const int row = mr + r + 8 * half;
                float z = fast_sigmoid(acc[nt][r] + bv);
                zs[nt][r] = z * bf16_to_f32(bufS[row * SP + col]);
            }
        }
        COMMIT();                                          // wts[p] = W(3)

        // ---- G;  omg = 1 - G (registers) ----
        v8f_t omg[8];
        zero_acc(acc);
        GSTEP(bufX, WW(4)); COMMIT();
        GSTEP(bufX, WW(5)); COMMIT();
        GSTEP(bufS, WW(9));
#pragma unroll
        for (int nt = 0; nt < 8; ++nt) {
            const int col = nt * 16 + ml;
            const float bv = BB(3)[col] + BB(4)[col] + BB(5)[col];
#pragma unroll
            for (int r = 0; r < 8; ++r)
                omg[nt][r] = 1.0f - fast_sigmoid(acc[nt][r] + bv);
        }
        COMMIT();                                          // wts[p] = W(9)

        // ---- H = tanh(x@w9 + (S*R)@w10 + S1@w11 + b);  S = omg*H + zs ----
        zero_acc(acc);
        GSTEP(bufX, WW(10)); COMMIT();
        GSTEP(bufT, WW(11)); COMMIT();
        GSTEP(bufX, (l < NDGM - 1) ? (WW(6) + (size_t)12 * HID * HID) : bw.h1T);
#pragma unroll
        for (int nt = 0; nt < 8; ++nt) {
            const int col = nt * 16 + ml;
            const float bv = BB(9)[col] + BB(10)[col] + BB(11)[col];
#pragma unroll
            for (int r = 0; r < 8; ++r) {
                const int row = mr + r + 8 * half;
                float h  = fast_tanh(acc[nt][r] + bv);
                float sn = omg[nt][r] * h + zs[nt][r];
                bufS[row * SP + col] = f32_to_bf16(sn);
            }
        }
        COMMIT();                                          // wts[p] = W(l+1,6)/dummy
#undef WW
#undef BB
#undef GSTEP
#undef COMMIT
    }

    __syncthreads();

    // --- output layer + hard constraints (fp32 VALU; tiny) ---
    if (tid < MT) {
        const int row  = tid;
        const int grow = R0 + row;
        if (bw.odim == 1) {
            float a = bw.out_b[0];
            for (int k = 0; k < HID; ++k)
                a += bf16_to_f32(bufS[row * SP + k]) * bw.out_w[k];
            float tv = T[grow];
            float ic = 1.0f;
            const float* xr = X + (size_t)grow * 16;
#pragma unroll
            for (int d = 0; d < 16; ++d) ic *= __sinf(3.14159265358979323846f * xr[d]);
            out[grow] = a * tv + ic;                       // u
        } else {
            float a[16];
#pragma unroll
            for (int j = 0; j < 16; ++j) a[j] = bw.out_b[j];
            for (int k = 0; k < HID; ++k) {
                const float s = bf16_to_f32(bufS[row * SP + k]);
                const float4* wr = (const float4*)(bw.out_w + (size_t)k * 16);
#pragma unroll
                for (int q = 0; q < 4; ++q) {
                    float4 w = wr[q];
                    a[4 * q + 0] += s * w.x;
                    a[4 * q + 1] += s * w.y;
                    a[4 * q + 2] += s * w.z;
                    a[4 * q + 3] += s * w.w;
                }
            }
            const float* xr = X + (size_t)grow * 16;
            float* po = out + (size_t)n + (size_t)grow * 16;
#pragma unroll
            for (int j = 0; j < 16; ++j) {
                float x = xr[j];
                po[j] = x * (1.0f - x) * a[j];             // p
            }
        }
    }
}

// ------------------------- host launch -------------------------------------

extern "C" void kernel_launch(void* const* d_in, const int* in_sizes, int n_in,
                              void* d_out, int out_size, void* d_ws, size_t ws_size,
                              hipStream_t stream) {
    const int N = in_sizes[0] / 16;   // X is (N, 16)

    const float* X = (const float*)d_in[0];
    const float* T = (const float*)d_in[1];

    // param dict order per branch: in_w,in_b,h1_w,h1_b,h2_w,h2_b,dgm_w,dgm_b,out_w,out_b
    const float* up[10];
    const float* pp[10];
    for (int i = 0; i < 10; ++i) { up[i] = (const float*)d_in[2 + i];
                                   pp[i] = (const float*)d_in[12 + i]; }

    // ---- workspace layout ----
    unsigned short* inp = (unsigned short*)d_ws;                   // N*32 bf16
    size_t off = (size_t)N * KIN * sizeof(unsigned short);
    off = (off + 255) & ~(size_t)255;

    const size_t sz_in  = (size_t)HID * KIN;          // 128*32
    const size_t sz_sq  = (size_t)HID * HID;          // 128*128
    const size_t sz_dgm = (size_t)NDGM * 12 * HID * HID;
    const size_t branch_elems = sz_in + 2 * sz_sq + sz_dgm;

    unsigned short* wb[2];
    wb[0] = (unsigned short*)((char*)d_ws + off);
    wb[1] = wb[0] + branch_elems;

    // ---- prep: pack input, convert/transpose weights to bf16 ----
    k_pack_input<<<(N + 255) / 256, 256, 0, stream>>>(X, T, inp, N);

    BranchW bws[2];
    const float** prm[2] = { up, pp };
    for (int b = 0; b < 2; ++b) {
        unsigned short* in_wT = wb[b];
        unsigned short* h1T   = in_wT + sz_in;
        unsigned short* h2T   = h1T + sz_sq;
        unsigned short* dgmT  = h2T + sz_sq;

        k_conv_pad<<<((int)sz_in + 255) / 256, 256, 0, stream>>>(prm[b][0], in_wT, DIN, HID, KIN);
        k_conv_sq<<<((int)sz_sq + 255) / 256, 256, 0, stream>>>(prm[b][2], h1T, 1);
        k_conv_sq<<<((int)sz_sq + 255) / 256, 256, 0, stream>>>(prm[b][4], h2T, 1);
        k_conv_sq<<<((int)sz_dgm + 255) / 256, 256, 0, stream>>>(prm[b][6], dgmT, NDGM * 12);

        bws[b].in_wT = in_wT;  bws[b].h1T = h1T;  bws[b].h2T = h2T;  bws[b].dgmT = dgmT;
        bws[b].in_b  = prm[b][1];
        bws[b].h1_b  = prm[b][3];
        bws[b].h2_b  = prm[b][5];
        bws[b].dgm_b = prm[b][7];
        bws[b].out_w = prm[b][8];
        bws[b].out_b = prm[b][9];
        bws[b].odim  = (b == 0) ? 1 : 16;
    }

    // ---- main fused kernel: (N/128) tiles x 2 branches ----
    dim3 grid(N / MT, 2);
    size_t lds_bytes = (size_t)5 * MT * SP * sizeof(unsigned short);   // 174,080 B
    dgm_branch_kernel<<<grid, 256, lds_bytes, stream>>>(
        X, T, inp, bws[0], bws[1], (float*)d_out, N);
}
